// PolicyCausalTransformer_66520453480556
// MI455X (gfx1250) — matrix operand order repaired and checked by reading, not compiled, not run on hardware
//
#include <hip/hip_runtime.h>

typedef __attribute__((ext_vector_type(16))) __bf16 v16bf;
typedef __attribute__((ext_vector_type(8)))  float  v8f;
typedef __attribute__((__vector_size__(16))) int    vi4;
typedef unsigned short u16;
typedef unsigned int   u32;

#define DEV __device__ __forceinline__

#if __has_builtin(__builtin_amdgcn_global_load_async_to_lds_b128)
#define USE_ASYNC 1
#endif

#define AS1 __attribute__((address_space(1)))
#define AS3 __attribute__((address_space(3)))

// ---- problem constants ----
constexpr int D    = 1024;
constexpr int H    = 16;
constexpr int HD   = 64;
constexpr int L    = 4;
constexpr int T    = 128;
constexpr int B    = 2;
constexpr int ONE  = 12;           // 8 img + 1 think + 1 ao + 2 act
constexpr int TTA  = 9;
constexpr int S    = ONE * T;      // 1536
constexpr int BS   = B * S;        // 3072
constexpr int CH   = 256;
constexpr int LDEC = 2;
constexpr int DH   = 8;
constexpr int DHD  = 128;
constexpr int BT   = B * T;        // 256
constexpr int ZR   = BT * 3;       // 768 decoder rows

// ---- helpers ----
DEV u16 f2bf(float f) {
  u32 u = __float_as_uint(f);
  u32 r = u + 0x7fffu + ((u >> 16) & 1u);   // round-to-nearest-even
  return (u16)(r >> 16);
}
DEV v16bf mkfrag(uint4 a, uint4 b) {
  union { uint4 q[2]; v16bf v; } u;
  u.q[0] = a; u.q[1] = b;
  return u.v;
}
DEV v16bf ld_frag2(const u16* p0, const u16* p1) {   // two 16B chunks
  return mkfrag(*(const uint4*)p0, *(const uint4*)p1);
}
DEV v16bf ld_frag1(const u16* p) {                   // contiguous 32B
  const uint4* q = (const uint4*)p;
  return mkfrag(q[0], q[1]);
}
DEV float gelu_t(float x) {
  float c = 0.7978845608028654f * (x + 0.044715f * x * x * x);
  return 0.5f * x * (1.0f + tanhf(c));
}
DEV void stv(float* p, float v) { *p = v; }
DEV void stv(u16*   p, float v) { *p = f2bf(v); }

// 16B global -> LDS copy, async (CDNA5 GLOBAL_LOAD_ASYNC_TO_LDS_B128) if available
DEV void cp16_g2l(const u16* g, u16* l) {
#ifdef USE_ASYNC
  __builtin_amdgcn_global_load_async_to_lds_b128((AS1 vi4*)g, (AS3 vi4*)l, 0, 0);
#else
  *(uint4*)l = *(const uint4*)g;
#endif
}
DEV void async_wait_all() {
#ifdef USE_ASYNC
  asm volatile("s_wait_asynccnt 0x0" ::: "memory");
#endif
}

// =====================================================================
// bf16 WMMA GEMM:  C(MxN) = A(MxK, bf16 row-major) * Bt^T
//   Bt is the weight stored transposed: [N x K] bf16 row-major.
// Block tile 128x128, K-step 64, double-buffered async LDS staging.
// Wave tile 32x64 -> 16 WMMA per K-step per wave.
// EPI 0: Cf=acc   1: Cf=acc+Res   2: Cb=bf16(gelu(acc))
// Requires M%128==0, N%128==0, K%64==0.
// =====================================================================
constexpr int KST = 64;   // K-step
constexpr int RST = 72;   // LDS row stride (elems): 64 + 8 pad, keeps 16B align

template<int EPI>
__global__ __launch_bounds__(256)
void gemm_wmma(const u16* __restrict__ Ab, const u16* __restrict__ Bt,
               float* __restrict__ Cf, u16* __restrict__ Cb,
               const float* __restrict__ Res, int M, int N, int K)
{
  __shared__ __align__(16) u16 lA[2][128 * RST];
  __shared__ __align__(16) u16 lB[2][128 * RST];

  int tid  = threadIdx.x;
  int lane = tid & 31, wv = tid >> 5;
  int wm = wv >> 1, wn = wv & 1;
  int m0 = blockIdx.x * 128;
  int n0 = blockIdx.y * 128;
  int mr = lane & 15, hi = lane >> 4;

  v8f acc[2][4];
#pragma unroll
  for (int i = 0; i < 2; i++)
#pragma unroll
    for (int c = 0; c < 4; c++) acc[i][c] = (v8f){0,0,0,0,0,0,0,0};

  // staging: A-tile 128x64 (1024 16B-chunks) + B-tile 128x64 -> 8 chunks/thread
  auto stage = [&](int buf, int k0) {
#pragma unroll
    for (int i = 0; i < 4; i++) {
      int c = tid + i * 256;
      int row = c >> 3, q = c & 7;
      cp16_g2l(Ab + (size_t)(m0 + row) * K + k0 + q * 8,
               &lA[buf][row * RST + q * 8]);
      cp16_g2l(Bt + (size_t)(n0 + row) * K + k0 + q * 8,
               &lB[buf][row * RST + q * 8]);
    }
  };

  stage(0, 0);
  int nk = K / KST, cur = 0;
  for (int kt = 0; kt < nk; kt++) {
    async_wait_all();
    __syncthreads();
    if (kt + 1 < nk) stage(cur ^ 1, (kt + 1) * KST);

    const u16* As = lA[cur];
    const u16* Bs = lB[cur];
    v16bf a[2][2];
#pragma unroll
    for (int i = 0; i < 2; i++) {
      const u16* ar = As + (wm * 32 + i * 16 + mr) * RST + hi * 8;
      a[i][0] = ld_frag2(ar,      ar + 16);   // k 0..31
      a[i][1] = ld_frag2(ar + 32, ar + 48);   // k 32..63
    }
#pragma unroll
    for (int c = 0; c < 4; c++) {
      const u16* bp = Bs + (wn * 64 + c * 16 + mr) * RST + hi * 8;
      v16bf b0 = ld_frag2(bp, bp + 16);        // k 0..31
      v16bf b1 = ld_frag2(bp + 32, bp + 48);   // k 32..63
#pragma unroll
      for (int i = 0; i < 2; i++) {
        acc[i][c] = __builtin_amdgcn_wmma_f32_16x16x32_bf16(
            false, a[i][0], false, b0, (short)0, acc[i][c], false, false);
        acc[i][c] = __builtin_amdgcn_wmma_f32_16x16x32_bf16(
            false, a[i][1], false, b1, (short)0, acc[i][c], false, false);
      }
    }
    cur ^= 1;
  }

  int mb = hi * 8;
#pragma unroll
  for (int i = 0; i < 2; i++)
#pragma unroll
    for (int c = 0; c < 4; c++)
#pragma unroll
      for (int r = 0; r < 8; r++) {
        size_t idx = (size_t)(m0 + wm * 32 + i * 16 + mb + r) * N +
                     n0 + wn * 64 + c * 16 + mr;
        float v = acc[i][c][r];
        if (EPI == 1) v += Res[idx];
        if (EPI == 2) Cb[idx] = f2bf(gelu_t(v));
        else          Cf[idx] = v;
      }
}

// =====================================================================
// Flash attention (encoder): per (qblock, h, b); 8 waves x 16 query rows.
// Q pre-scaled by 1/8. K row-major [S x 64], V transposed [64 x S], bf16.
// =====================================================================
__global__ __launch_bounds__(256)
void flash_attn(const u16* __restrict__ Q, const u16* __restrict__ Kb,
                const u16* __restrict__ Vt, u16* __restrict__ O)
{
  __shared__ __align__(16) u16 pl[8][16 * 32];
  int lane = threadIdx.x & 31, wv = threadIdx.x >> 5;
  int b = blockIdx.z, h = blockIdx.y;
  int q0 = blockIdx.x * 128 + wv * 16;
  int mr = lane & 15, hi = lane >> 4;
  int nc = mr, mb = hi * 8;                 // C-layout: lane half -> rows mb..mb+7

  size_t hq = ((size_t)(b * H + h)) * S * 64;
  const u16* Qp = Q + hq;
  const u16* Kp = Kb + hq;
  const u16* Vp = Vt + ((size_t)(b * H + h)) * 64 * S;

  const u16* qr = Qp + (size_t)(q0 + mr) * 64 + hi * 8;
  v16bf qa0 = ld_frag2(qr, qr + 16);        // d 0..31
  v16bf qa1 = ld_frag2(qr + 32, qr + 48);   // d 32..63

  v8f o[4];
#pragma unroll
  for (int c = 0; c < 4; c++) o[c] = (v8f){0,0,0,0,0,0,0,0};
  float mv[8], lv[8];
#pragma unroll
  for (int r = 0; r < 8; r++) { mv[r] = -3e38f; lv[r] = 0.f; }

  int kmax = ((q0 + 15) / ONE + 1) * ONE;   // exclusive; <= S (S%32==0)
  int nkt = (kmax + 31) / 32;
  u16* P = pl[wv];

  for (int kt = 0; kt < nkt; kt++) {
    int k0 = kt * 32;
    v8f s[2];
#pragma unroll
    for (int j = 0; j < 2; j++) {
      const u16* kr = Kp + (size_t)(k0 + j * 16 + mr) * 64 + hi * 8;
      v8f z = (v8f){0,0,0,0,0,0,0,0};
      z = __builtin_amdgcn_wmma_f32_16x16x32_bf16(false, qa0, false, ld_frag2(kr, kr + 16),      (short)0, z, false, false);
      z = __builtin_amdgcn_wmma_f32_16x16x32_bf16(false, qa1, false, ld_frag2(kr + 32, kr + 48), (short)0, z, false, false);
      s[j] = z;
    }
    // mask (HIST window always satisfied since T==HIST)
#pragma unroll
    for (int j = 0; j < 2; j++) {
      int k = k0 + j * 16 + nc;
      int ks = k / ONE, kp = k - ks * ONE;
#pragma unroll
      for (int r = 0; r < 8; r++) {
        int q = q0 + mb + r;
        int qs = q / ONE, qp = q - qs * ONE;
        bool allowed;
        if (ks < qs)        allowed = (kp != TTA);
        else if (ks == qs) {
          if (qp < TTA)       allowed = (kp < TTA);
          else if (qp == TTA) allowed = (kp <= TTA);
          else                allowed = (kp != TTA);
        } else               allowed = false;
        if (!allowed) s[j][r] = -1e9f;
      }
    }
    // online softmax (row stats per lane-half; vgpr r == row mb+r)
    float rmax[8];
#pragma unroll
    for (int r = 0; r < 8; r++) rmax[r] = fmaxf(s[0][r], s[1][r]);
#pragma unroll
    for (int m = 1; m < 16; m <<= 1)
#pragma unroll
      for (int r = 0; r < 8; r++) rmax[r] = fmaxf(rmax[r], __shfl_xor(rmax[r], m, 32));
    float sc[8];
#pragma unroll
    for (int r = 0; r < 8; r++) {
      float mn = fmaxf(mv[r], rmax[r]);
      sc[r] = __expf(mv[r] - mn);
      mv[r] = mn;
    }
#pragma unroll
    for (int j = 0; j < 2; j++)
#pragma unroll
      for (int r = 0; r < 8; r++) s[j][r] = __expf(s[j][r] - mv[r]);
    float rs[8];
#pragma unroll
    for (int r = 0; r < 8; r++) rs[r] = s[0][r] + s[1][r];
#pragma unroll
    for (int m = 1; m < 16; m <<= 1)
#pragma unroll
      for (int r = 0; r < 8; r++) rs[r] += __shfl_xor(rs[r], m, 32);
#pragma unroll
    for (int r = 0; r < 8; r++) lv[r] = lv[r] * sc[r] + rs[r];
#pragma unroll
    for (int c = 0; c < 4; c++)
#pragma unroll
      for (int r = 0; r < 8; r++) o[c][r] *= sc[r];

    // P tile: C-layout -> LDS (row-major 16x32 bf16) -> A-layout fragment
#pragma unroll
    for (int j = 0; j < 2; j++)
#pragma unroll
      for (int r = 0; r < 8; r++) P[(mb + r) * 32 + j * 16 + nc] = f2bf(s[j][r]);
    asm volatile("s_wait_dscnt 0" ::: "memory");
    const u16* pr = P + mr * 32 + hi * 8;
    v16bf pa = ld_frag2(pr, pr + 16);

#pragma unroll
    for (int c = 0; c < 4; c++) {
      const u16* vr = Vp + (size_t)(c * 16 + mr) * S + k0 + hi * 8;
      o[c] = __builtin_amdgcn_wmma_f32_16x16x32_bf16(
          false, pa, false, ld_frag2(vr, vr + 16), (short)0, o[c], false, false);
    }
  }
  float inv[8];
#pragma unroll
  for (int r = 0; r < 8; r++) inv[r] = 1.0f / lv[r];
#pragma unroll
  for (int c = 0; c < 4; c++)
#pragma unroll
    for (int r = 0; r < 8; r++)
      O[(size_t)(b * S + q0 + mb + r) * D + h * 64 + c * 16 + nc] = f2bf(o[c][r] * inv[r]);
}

// =====================================================================
// LayerNorm over rows of D=1024; generic row/batch strides; bf16 or f32 out.
// =====================================================================
template<typename OT>
__global__ __launch_bounds__(256)
void ln_rows(const float* __restrict__ in, size_t irs, size_t ibs,
             OT* __restrict__ out, size_t ors, size_t obs,
             const float* __restrict__ w, const float* __restrict__ bia)
{
  const float* x = in + (size_t)blockIdx.y * ibs + (size_t)blockIdx.x * irs;
  OT* y = out + (size_t)blockIdx.y * obs + (size_t)blockIdx.x * ors;
  float v[4], s = 0.f, s2 = 0.f;
#pragma unroll
  for (int i = 0; i < 4; i++) {
    v[i] = x[threadIdx.x + i * 256];
    s += v[i]; s2 += v[i] * v[i];
  }
#pragma unroll
  for (int m = 1; m < 32; m <<= 1) { s += __shfl_xor(s, m, 32); s2 += __shfl_xor(s2, m, 32); }
  __shared__ float rb[16];
  int wv = threadIdx.x >> 5;
  if ((threadIdx.x & 31) == 0) { rb[wv] = s; rb[8 + wv] = s2; }
  __syncthreads();
  s = 0.f; s2 = 0.f;
#pragma unroll
  for (int i = 0; i < 8; i++) { s += rb[i]; s2 += rb[8 + i]; }
  float mean = s * (1.f / 1024.f);
  float var  = s2 * (1.f / 1024.f) - mean * mean;
  float rstd = rsqrtf(var + 1e-5f);
#pragma unroll
  for (int i = 0; i < 4; i++) {
    int d = threadIdx.x + i * 256;
    stv(&y[d], (v[i] - mean) * rstd * w[d] + bia[d]);
  }
}

// =====================================================================
// Token embedding: img GEMM(256) + pos, think/ao broadcast, act + pos. f32 out.
// =====================================================================
__global__ __launch_bounds__(256)
void embed(const float* __restrict__ ram, const float* __restrict__ aei,
           const float* __restrict__ Wr,  const float* __restrict__ ipos,
           const float* __restrict__ tpos, const float* __restrict__ aotok,
           const float* __restrict__ apos, float* __restrict__ x)
{
  int row = blockIdx.x;
  int b = row / S, s = row - b * S;
  int t = s / ONE, p = s - t * ONE;
  float* xr = x + (size_t)row * D;
  if (p < 8) {
    __shared__ float rsh[CH];
    const float* rp = ram + ((size_t)(b * T + t)) * 2048 + p * CH;
    rsh[threadIdx.x] = rp[threadIdx.x];
    __syncthreads();
    float a[4] = {0.f, 0.f, 0.f, 0.f};
    for (int c = 0; c < CH; c++) {
      float rv = rsh[c];
      const float* wrow = Wr + (size_t)c * D + threadIdx.x;
#pragma unroll
      for (int i = 0; i < 4; i++) a[i] += rv * wrow[i * 256];
    }
#pragma unroll
    for (int i = 0; i < 4; i++) {
      int d = threadIdx.x + i * 256;
      xr[d] = a[i] + ipos[p * D + d];
    }
  } else if (p == 8) {
#pragma unroll
    for (int i = 0; i < 4; i++) { int d = threadIdx.x + i * 256; xr[d] = tpos[d]; }
  } else if (p == 9) {
#pragma unroll
    for (int i = 0; i < 4; i++) { int d = threadIdx.x + i * 256; xr[d] = aotok[d]; }
  } else {
    int j = p - 10;
#pragma unroll
    for (int i = 0; i < 4; i++) {
      int d = threadIdx.x + i * 256;
      xr[d] = aei[((size_t)(b * T + t) * 2 + j) * D + d] + apos[j * D + d];
    }
  }
}

// =====================================================================
// qkv (f32, [BS x 3D]) -> per-head bf16 Q (pre-scaled 1/8), K [S x 64], V^T [64 x S]
// =====================================================================
__global__ __launch_bounds__(256)
void qkv_rearrange(const float* __restrict__ qkv, u16* __restrict__ Q,
                   u16* __restrict__ K, u16* __restrict__ Vt)
{
  int row = blockIdx.x;
  int b = row / S, s = row - b * S;
  const float* qr = qkv + (size_t)row * (3 * D);
#pragma unroll
  for (int i = 0; i < 4; i++) {
    int col = threadIdx.x + i * 256;
    int h = col >> 6, hd = col & 63;
    size_t hb = (size_t)(b * H + h);
    Q[(hb * S + s) * 64 + hd]  = f2bf(qr[col] * 0.125f);
    K[(hb * S + s) * 64 + hd]  = f2bf(qr[D + col]);
    Vt[(hb * 64 + hd) * S + s] = f2bf(qr[2 * D + col]);
  }
}

// weight f32 [K x N] -> bf16 transposed [N x K]
__global__ __launch_bounds__(256)
void convT(const float* __restrict__ W, u16* __restrict__ Wt, int K, int N)
{
  size_t i = (size_t)blockIdx.x * 256 + threadIdx.x;  // K*N % 256 == 0
  int n = (int)(i / K);
  int k = (int)(i - (size_t)n * K);
  Wt[i] = f2bf(W[(size_t)k * N + n]);
}

// z rows 1,2 <- raw action embeddings
__global__ __launch_bounds__(256)
void fill_zact(const float* __restrict__ aei, float* __restrict__ z)
{
  size_t i = (size_t)blockIdx.x * 256 + threadIdx.x;  // BT*2*D total
  int d = (int)(i & 1023);
  size_t rj = i >> 10;
  int j  = (int)(rj & 1);
  int bt = (int)(rj >> 1);
  z[((size_t)bt * 3 + 1 + j) * D + d] = aei[((size_t)bt * 2 + j) * D + d];
}

// =====================================================================
// Decoder attention: seq 3, 8 heads x 128 dim, full mask. f32 in -> bf16 out.
// =====================================================================
__global__ __launch_bounds__(128)
void dec_attn(const float* __restrict__ qkv, u16* __restrict__ ob)
{
  int bt = blockIdx.x, h = blockIdx.y, d = threadIdx.x;
  int lane = threadIdx.x & 31, wv = threadIdx.x >> 5;
  const float* base = qkv + (size_t)bt * 3 * (3 * D) + h * DHD + d;
  float q[3], k[3], v[3];
#pragma unroll
  for (int i = 0; i < 3; i++) {
    q[i] = base[(size_t)i * (3 * D)];
    k[i] = base[(size_t)i * (3 * D) + D];
    v[i] = base[(size_t)i * (3 * D) + 2 * D];
  }
  __shared__ float sred[9][4];
#pragma unroll
  for (int i = 0; i < 3; i++)
#pragma unroll
    for (int j = 0; j < 3; j++) {
      float p = q[i] * k[j];
#pragma unroll
      for (int m = 1; m < 32; m <<= 1) p += __shfl_xor(p, m, 32);
      if (lane == 0) sred[i * 3 + j][wv] = p;
    }
  __syncthreads();
  float sc[3][3];
#pragma unroll
  for (int i = 0; i < 3; i++)
#pragma unroll
    for (int j = 0; j < 3; j++)
      sc[i][j] = (sred[i * 3 + j][0] + sred[i * 3 + j][1] +
                  sred[i * 3 + j][2] + sred[i * 3 + j][3]) * 0.08838834764831845f;
#pragma unroll
  for (int i = 0; i < 3; i++) {
    float mx = fmaxf(sc[i][0], fmaxf(sc[i][1], sc[i][2]));
    float e0 = __expf(sc[i][0] - mx), e1 = __expf(sc[i][1] - mx), e2 = __expf(sc[i][2] - mx);
    float inv = 1.0f / (e0 + e1 + e2);
    float o = (e0 * v[0] + e1 * v[1] + e2 * v[2]) * inv;
    ob[((size_t)(bt * 3 + i)) * D + h * DHD + d] = f2bf(o);
  }
}

// =====================================================================
extern "C" void kernel_launch(void* const* d_in, const int* in_sizes, int n_in,
                              void* d_out, int out_size, void* d_ws, size_t ws_size,
                              hipStream_t stream)
{
  (void)in_sizes; (void)n_in; (void)out_size; (void)ws_size;
  const float* ram   = (const float*)d_in[0];
  const float* aei   = (const float*)d_in[1];
  const float* W_ram = (const float*)d_in[2];
  const float* ipos  = (const float*)d_in[3];
  const float* tpos  = (const float*)d_in[4];
  const float* aotok = (const float*)d_in[5];
  const float* apos  = (const float*)d_in[6];
  const float* ln1w  = (const float*)d_in[7];
  const float* ln1b  = (const float*)d_in[8];
  const float* Wqkv  = (const float*)d_in[9];
  const float* Wo    = (const float*)d_in[10];
  const float* ln2w  = (const float*)d_in[11];
  const float* ln2b  = (const float*)d_in[12];
  const float* Wm1   = (const float*)d_in[13];
  const float* Wm2   = (const float*)d_in[14];
  const float* lnfw  = (const float*)d_in[15];
  const float* lnfb  = (const float*)d_in[16];
  const float* dln1w = (const float*)d_in[17];
  const float* dln1b = (const float*)d_in[18];
  const float* dWqkv = (const float*)d_in[19];
  const float* dWo   = (const float*)d_in[20];
  const float* dln2w = (const float*)d_in[21];
  const float* dln2b = (const float*)d_in[22];
  const float* dWm1  = (const float*)d_in[23];
  const float* dWm2  = (const float*)d_in[24];
  const float* dlnfw = (const float*)d_in[25];
  const float* dlnfb = (const float*)d_in[26];
  float* out = (float*)d_out;

  char* ws = (char*)d_ws;
  size_t off = 0;
  auto alloc = [&](size_t bytes) -> char* {
    char* p = ws + off;
    off = (off + bytes + 255) & ~(size_t)255;
    return p;
  };
  float* x    = (float*)alloc((size_t)BS * D * 4);
  u16*   xn   = (u16*)  alloc((size_t)BS * D * 2);
  float* qkv  = (float*)alloc((size_t)BS * 3 * D * 4);
  u16*   qb   = (u16*)  alloc((size_t)B * H * S * 64 * 2);
  u16*   kb   = (u16*)  alloc((size_t)B * H * S * 64 * 2);
  u16*   vt   = (u16*)  alloc((size_t)B * H * 64 * S * 2);
  u16*   ob   = (u16*)  alloc((size_t)BS * D * 2);
  u16*   h1   = (u16*)  alloc((size_t)BS * 4 * D * 2);
  u16*   wt   = (u16*)  alloc((size_t)4 * D * D * 2);     // largest weight (1024x4096)
  float* z    = (float*)alloc((size_t)ZR * D * 4);
  u16*   zn   = (u16*)  alloc((size_t)ZR * D * 2);
  float* dqkv = (float*)alloc((size_t)ZR * 3 * D * 4);
  u16*   dob  = (u16*)  alloc((size_t)ZR * D * 2);
  u16*   dh1  = (u16*)  alloc((size_t)ZR * 4 * D * 2);

  // ---- embed ----
  embed<<<BS, 256, 0, stream>>>(ram, aei, W_ram, ipos, tpos, aotok, apos, x);

  // ---- encoder layers ----
  for (int i = 0; i < L; i++) {
    ln_rows<u16><<<dim3(BS, 1), 256, 0, stream>>>(
        x, D, 0, xn, D, 0, ln1w + i * D, ln1b + i * D);

    convT<<<(D * 3 * D) / 256, 256, 0, stream>>>(Wqkv + (size_t)i * D * 3 * D, wt, D, 3 * D);
    gemm_wmma<0><<<dim3(BS / 128, (3 * D) / 128), 256, 0, stream>>>(
        xn, wt, qkv, nullptr, nullptr, BS, 3 * D, D);

    qkv_rearrange<<<BS, 256, 0, stream>>>(qkv, qb, kb, vt);
    flash_attn<<<dim3(S / 128, H, B), 256, 0, stream>>>(qb, kb, vt, ob);

    convT<<<(D * D) / 256, 256, 0, stream>>>(Wo + (size_t)i * D * D, wt, D, D);
    gemm_wmma<1><<<dim3(BS / 128, D / 128), 256, 0, stream>>>(
        ob, wt, x, nullptr, x, BS, D, D);

    ln_rows<u16><<<dim3(BS, 1), 256, 0, stream>>>(
        x, D, 0, xn, D, 0, ln2w + i * D, ln2b + i * D);

    convT<<<(D * 4 * D) / 256, 256, 0, stream>>>(Wm1 + (size_t)i * D * 4 * D, wt, D, 4 * D);
    gemm_wmma<2><<<dim3(BS / 128, (4 * D) / 128), 256, 0, stream>>>(
        xn, wt, nullptr, h1, nullptr, BS, 4 * D, D);

    convT<<<(4 * D * D) / 256, 256, 0, stream>>>(Wm2 + (size_t)i * 4 * D * D, wt, 4 * D, D);
    gemm_wmma<1><<<dim3(BS / 128, D / 128), 256, 0, stream>>>(
        h1, wt, x, nullptr, x, BS, D, 4 * D);
  }

  // ---- build decoder input z: row0 = lnf(ao), rows1,2 = raw action embeds ----
  ln_rows<float><<<dim3(T, B), 256, 0, stream>>>(
      x + (size_t)TTA * D, (size_t)ONE * D, (size_t)S * D,
      z, (size_t)3 * D, (size_t)T * 3 * D, lnfw, lnfb);
  fill_zact<<<(BT * 2 * D) / 256, 256, 0, stream>>>(aei, z);

  // ---- decoder layers ----
  for (int i = 0; i < LDEC; i++) {
    ln_rows<u16><<<dim3(ZR, 1), 256, 0, stream>>>(
        z, D, 0, zn, D, 0, dln1w + i * D, dln1b + i * D);

    convT<<<(D * 3 * D) / 256, 256, 0, stream>>>(dWqkv + (size_t)i * D * 3 * D, wt, D, 3 * D);
    gemm_wmma<0><<<dim3(ZR / 128, (3 * D) / 128), 256, 0, stream>>>(
        zn, wt, dqkv, nullptr, nullptr, ZR, 3 * D, D);

    dec_attn<<<dim3(BT, DH), 128, 0, stream>>>(dqkv, dob);

    convT<<<(D * D) / 256, 256, 0, stream>>>(dWo + (size_t)i * D * D, wt, D, D);
    gemm_wmma<1><<<dim3(ZR / 128, D / 128), 256, 0, stream>>>(
        dob, wt, z, nullptr, z, ZR, D, D);

    ln_rows<u16><<<dim3(ZR, 1), 256, 0, stream>>>(
        z, D, 0, zn, D, 0, dln2w + i * D, dln2b + i * D);

    convT<<<(D * 4 * D) / 256, 256, 0, stream>>>(dWm1 + (size_t)i * D * 4 * D, wt, D, 4 * D);
    gemm_wmma<2><<<dim3(ZR / 128, (4 * D) / 128), 256, 0, stream>>>(
        zn, wt, nullptr, dh1, nullptr, ZR, 4 * D, D);

    convT<<<(4 * D * D) / 256, 256, 0, stream>>>(dWm2 + (size_t)i * 4 * D * D, wt, 4 * D, D);
    gemm_wmma<1><<<dim3(ZR / 128, D / 128), 256, 0, stream>>>(
        dh1, wt, z, nullptr, z, ZR, D, 4 * D);
  }

  // ---- final LN -> output (B,T,3,D) f32 ----
  ln_rows<float><<<dim3(ZR, 1), 256, 0, stream>>>(
      z, D, 0, out, D, 0, dlnfw, dlnfb);
}
